// LaSAttention_2972117369221
// MI455X (gfx1250) — matrix operand order repaired and checked by reading, not compiled, not run on hardware
//
#include <hip/hip_runtime.h>

typedef __attribute__((ext_vector_type(16))) _Float16 v16h;
typedef __attribute__((ext_vector_type(8)))  _Float16 h8;
typedef __attribute__((ext_vector_type(4)))  _Float16 h4;
typedef __attribute__((ext_vector_type(8)))  float    v8f;
typedef __attribute__((ext_vector_type(4)))  float    f4;

#define S_LEN    2048
#define DHEAD    64
#define ROWS     16      // query rows per workgroup
#define KBLK     128     // key rows staged in LDS per iteration
#define NTHREADS 256
#define SEGW     (S_LEN / 16)   // 128 keys per softmax thread

// dynamic LDS layout (bytes)
#define OFF_SCORES 0
#define OFF_PF16   (ROWS * S_LEN * 4)                 // 131072
#define OFF_KTILE  (OFF_PF16 + ROWS * S_LEN * 2)      // 196608
#define OFF_QT     (OFF_KTILE + KBLK * DHEAD * 2)     // 212992
#define OFF_RED    (OFF_QT + ROWS * DHEAD * 2)        // 215040
#define OFF_OACC   (OFF_RED + NTHREADS * 4)           // 216064
#define SMEM_BYTES (OFF_OACC + 2 * ROWS * DHEAD * 4)  // 224256

#define CONCAT16(lo, hi) __builtin_shufflevector(lo, hi, 0,1,2,3,4,5,6,7,8,9,10,11,12,13,14,15)

__global__ __launch_bounds__(NTHREADS)
void las_attn_wmma_kernel(const float* __restrict__ q,
                          const float* __restrict__ k,
                          const float* __restrict__ v,
                          float* __restrict__ outO_base,   // [BH, S, D]
                          float* __restrict__ outP_base)   // [BH, S, S]
{
    extern __shared__ char smem[];
    float*    sc   = (float*)(smem + OFF_SCORES);    // [16][2048] scores / exp vals
    _Float16* pf   = (_Float16*)(smem + OFF_PF16);   // [16][2048] pooled probs f16
    _Float16* kt   = (_Float16*)(smem + OFF_KTILE);  // [128][64]  K block f16
    _Float16* qt   = (_Float16*)(smem + OFF_QT);     // [16][64]   Q tile f16
    float*    red  = (float*)(smem + OFF_RED);       // [256] reduction scratch
    float*    oacc = (float*)(smem + OFF_OACC);      // [2][16][64] PV partials

    const int t    = threadIdx.x;
    const int lane = t & 31;
    const int wave = t >> 5;
    const int i0   = blockIdx.x * ROWS;
    const int bh   = blockIdx.y;

    const float* qg = q + (size_t)bh * S_LEN * DHEAD;
    const float* kg = k + (size_t)bh * S_LEN * DHEAD;
    const float* vg = v + (size_t)bh * S_LEN * DHEAD;
    float* outO = outO_base + (size_t)bh * S_LEN * DHEAD + (size_t)i0 * DHEAD;
    float* outP = outP_base + (size_t)bh * S_LEN * S_LEN;

    // ---- Phase 1: stage Q tile (f32 -> f16), vectorized --------------------
    for (int x = 4 * t; x < ROWS * DHEAD; x += 4 * NTHREADS) {
        const f4 f = *(const f4*)(qg + (size_t)i0 * DHEAD + x);
        h4 h = { (_Float16)f[0], (_Float16)f[1], (_Float16)f[2], (_Float16)f[3] };
        *(h4*)(qt + x) = h;
    }
    __syncthreads();

    // A fragments for Q (16x64 = 2 chunks of 16x32), built once per wave.
    // A layout: row = lane%16; K = e + (e>=8 ? 8:0) + (lane>=16 ? 8:0)
    //   -> two contiguous 8-half runs at offsets offA and offA+16.
    const int offA = (lane >= 16) ? 8 : 0;
    v16h aq0, aq1;
    {
        const _Float16* qrow = qt + (lane & 15) * DHEAD;
        h8 l0 = *(const h8*)(qrow + offA);
        h8 u0 = *(const h8*)(qrow + offA + 16);
        h8 l1 = *(const h8*)(qrow + 32 + offA);
        h8 u1 = *(const h8*)(qrow + 32 + offA + 16);
        aq0 = CONCAT16(l0, u0);
        aq1 = CONCAT16(l1, u1);
    }

    // ---- Phase 2: scores = (q/64 . k) * (-alpha*(j-i)) into LDS ------------
    const int offB = (lane >= 16) ? 8 : 0;
    for (int blk = 0; blk < S_LEN / KBLK; ++blk) {
        const int j0 = blk * KBLK;
        __syncthreads();  // protect kt from readers of previous block
        for (int x = 4 * t; x < KBLK * DHEAD; x += 4 * NTHREADS) {
            const f4 f = *(const f4*)(kg + (size_t)j0 * DHEAD + x);
            h4 h = { (_Float16)f[0], (_Float16)f[1], (_Float16)f[2], (_Float16)f[3] };
            *(h4*)(kt + x) = h;
        }
        __syncthreads();

        // wave w computes the 16-column tile at j0 + w*16.
        // B layout: N = lane%16; K = e + (e>=8 ? 8:0) + (lane>=16 ? 8:0) -> two 8-half runs.
        const _Float16* krow = kt + (wave * 16 + (lane & 15)) * DHEAD + offB;
        v16h b0 = CONCAT16(*(const h8*)(krow), *(const h8*)(krow + 16));
        v16h b1 = CONCAT16(*(const h8*)(krow + 32), *(const h8*)(krow + 48));

        v8f acc = {};
        acc = __builtin_amdgcn_wmma_f32_16x16x32_f16(
            false, aq0, false, b0, (short)0, acc, false, false);
        acc = __builtin_amdgcn_wmma_f32_16x16x32_f16(
            false, aq1, false, b1, (short)0, acc, false, false);

        // scale + distance decay, spill to score strip
        const int jj = j0 + wave * 16 + (lane & 15);
#pragma unroll
        for (int r = 0; r < 8; ++r) {
            const int m = r + (lane >= 16 ? 8 : 0);
            const float eld = -0.01f * (float)(jj - (i0 + m));
            sc[m * S_LEN + jj] = acc[r] * (1.0f / 64.0f) * eld;
        }
    }
    __syncthreads();

    // ---- Phase 3: row softmax stats (16 threads per row) -------------------
    const int row = t >> 4;
    const int seg = t & 15;
    float* srow = sc + row * S_LEN;

    float lm = -3.402823466e38f;
    for (int x = 0; x < SEGW; x += 4) {
        const f4 c = *(const f4*)(srow + seg * SEGW + x);
        lm = fmaxf(lm, fmaxf(fmaxf(c[0], c[1]), fmaxf(c[2], c[3])));
    }
    red[t] = lm;
    __syncthreads();
    float rowmax = red[row * 16];
#pragma unroll
    for (int s = 1; s < 16; ++s) rowmax = fmaxf(rowmax, red[row * 16 + s]);
    __syncthreads();  // everyone done reading red before reuse

    float lsum = 0.0f;
    for (int x = 0; x < SEGW; x += 4) {
        float* p = srow + seg * SEGW + x;
        f4 c = *(const f4*)p;
        c[0] = __expf(c[0] - rowmax);
        c[1] = __expf(c[1] - rowmax);
        c[2] = __expf(c[2] - rowmax);
        c[3] = __expf(c[3] - rowmax);
        *(f4*)p = c;  // unnormalized exp back into LDS
        lsum += c[0] + c[1] + c[2] + c[3];
    }
    red[t] = lsum;
    __syncthreads();      // also fences all sc exp writes
    float rowsum = 0.0f;
#pragma unroll
    for (int s = 0; s < 16; ++s) rowsum += red[row * 16 + s];
    const float inv3 = (1.0f / 3.0f) / rowsum;

    // ---- Phase 4: fused normalize + 3-tap avg pool -------------------------
    // f32 -> global (nontemporal: 536MB stream, never re-read, keep K/V in L2)
    // f16 -> LDS for the PV WMMA
    float* outPr = outP + (size_t)(i0 + row) * S_LEN;
    _Float16* pfr = pf + row * S_LEN;
    for (int x = 0; x < SEGW; x += 4) {
        const int j = seg * SEGW + x;
        const float left  = (j > 0)             ? srow[j - 1] : 0.0f;
        const f4    c     = *(const f4*)(srow + j);
        const float right = (j + 4 < S_LEN)     ? srow[j + 4] : 0.0f;
        f4 pooled;
        pooled[0] = (left + c[0] + c[1]) * inv3;
        pooled[1] = (c[0] + c[1] + c[2]) * inv3;
        pooled[2] = (c[1] + c[2] + c[3]) * inv3;
        pooled[3] = (c[2] + c[3] + right) * inv3;
        __builtin_nontemporal_store(pooled, (f4*)(outPr + j));
        h4 ph = { (_Float16)pooled[0], (_Float16)pooled[1],
                  (_Float16)pooled[2], (_Float16)pooled[3] };
        *(h4*)(pfr + j) = ph;
    }
    __syncthreads();

    // ---- Phase 5: O = P_smooth @ V, split-K across waves -------------------
    // wave w: d-column tile nb = (w&3)*16, K half = (w>>2)*1024
    const int nb    = (wave & 3) * 16;
    const int khalf = wave >> 2;
    v8f oa = {};
    for (int kc = 0; kc < 32; ++kc) {
        const int k0 = khalf * 1024 + kc * 32;

        const _Float16* prow = pf + (lane & 15) * S_LEN + k0 + offA;
        h8 plo = *(const h8*)(prow);
        h8 phi = *(const h8*)(prow + 16);
        v16h am = CONCAT16(plo, phi);

        v16h bm;
#pragma unroll
        for (int e = 0; e < 16; ++e) {
            const int keb = (e & 7) + ((e >> 3) << 4) + (lane >= 16 ? 8 : 0);
            bm[e] = (_Float16)vg[(size_t)(k0 + keb) * DHEAD + nb + (lane & 15)];
        }
        oa = __builtin_amdgcn_wmma_f32_16x16x32_f16(
            false, am, false, bm, (short)0, oa, false, false);
    }
    {
        const int n = lane & 15;
#pragma unroll
        for (int r = 0; r < 8; ++r) {
            const int m = r + (lane >= 16 ? 8 : 0);
            oacc[khalf * ROWS * DHEAD + m * DHEAD + nb + n] = oa[r];
        }
    }
    __syncthreads();
    for (int x = 4 * t; x < ROWS * DHEAD; x += 4 * NTHREADS) {
        const f4 a = *(const f4*)(oacc + x);
        const f4 b = *(const f4*)(oacc + ROWS * DHEAD + x);
        f4 o;
        o[0] = a[0] + b[0]; o[1] = a[1] + b[1];
        o[2] = a[2] + b[2]; o[3] = a[3] + b[3];
        *(f4*)(outO + x) = o;
    }
}

extern "C" void kernel_launch(void* const* d_in, const int* in_sizes, int n_in,
                              void* d_out, int out_size, void* d_ws, size_t ws_size,
                              hipStream_t stream) {
    (void)n_in; (void)out_size; (void)d_ws; (void)ws_size;
    const float* q = (const float*)d_in[0];
    const float* k = (const float*)d_in[1];
    const float* v = (const float*)d_in[2];
    float* out = (float*)d_out;

    const int BH = in_sizes[0] / (S_LEN * DHEAD);      // B*H = 32
    float* outO = out;                                  // [BH,S,D]
    float* outP = out + (size_t)BH * S_LEN * DHEAD;     // [BH,S,S]

    dim3 grid(S_LEN / ROWS, BH);   // 128 x 32
    dim3 block(NTHREADS);
    las_attn_wmma_kernel<<<grid, block, SMEM_BYTES, stream>>>(q, k, v, outO, outP);
}
